// MultiHeadAttention_1022202217043
// MI455X (gfx1250) — compile-verified, matrix operand order checked by reading, not run on hardware
//
#include <hip/hip_runtime.h>
#include <hip/hip_bf16.h>
#include <stdint.h>

// MHA forward for MI455X (gfx1250, wave32, WMMA, async global->LDS copies).
// B=4, S=1024, D=1024, H=16, depth=64.
//
// Workspace (halves/bf16, 64 MB total):
//   xq,xk,xv bf16 [4096,1024]       : 0, 4M, 8M
//   wq,wk,wv,dense_w bf16 [1024,1024]: 12M, 13M, 14M, 15M
//   qh,kh head-split [B,H,S,64]     : 16M, 20M
//   vt transposed   [B*H, 64, S]    : 24M
//   ctx merged      [B,S,1024]      : 28M
// d_out: [0,4M) fp32 output ; [4M,68M) fp32 attention weights.

typedef __attribute__((ext_vector_type(16))) __bf16          v16bf;
typedef __attribute__((ext_vector_type(8)))  float           v8f;
typedef __attribute__((ext_vector_type(8)))  unsigned short  v8u16;
typedef __attribute__((ext_vector_type(16))) unsigned short  v16u16;

struct U16x16 { v8u16 lo, hi; };

#define WMMA_BF16(a, b, c) \
  __builtin_amdgcn_wmma_f32_16x16x32_bf16(false, (a), false, (b), (short)0, (c), false, false)

__device__ __forceinline__ unsigned short f2bfu(float f) {
  unsigned int u = __builtin_bit_cast(unsigned int, f);
  u += 0x7FFFu + ((u >> 16) & 1u);
  return (unsigned short)(u >> 16);
}

// Lane fragment per CDNA5 16-bit A/B layout: lanes 0-15 hold K={0..7,16..23},
// lanes 16-31 hold K={8..15,24..31}; kb = 8*(lane>=16).
__device__ __forceinline__ v16bf frag_u16(const unsigned short* rowp, int kb) {
  U16x16 f;
  f.lo = *(const v8u16*)(rowp + kb);
  f.hi = *(const v8u16*)(rowp + kb + 16);
  return __builtin_bit_cast(v16bf, f);
}

__device__ __forceinline__ v16bf frag_f32(const float* rowp, int kb) {
  float4 a0 = *(const float4*)(rowp + kb);
  float4 a1 = *(const float4*)(rowp + kb + 4);
  float4 b0 = *(const float4*)(rowp + kb + 16);
  float4 b1 = *(const float4*)(rowp + kb + 20);
  v16u16 r;
  r[0]  = f2bfu(a0.x); r[1]  = f2bfu(a0.y); r[2]  = f2bfu(a0.z); r[3]  = f2bfu(a0.w);
  r[4]  = f2bfu(a1.x); r[5]  = f2bfu(a1.y); r[6]  = f2bfu(a1.z); r[7]  = f2bfu(a1.w);
  r[8]  = f2bfu(b0.x); r[9]  = f2bfu(b0.y); r[10] = f2bfu(b0.z); r[11] = f2bfu(b0.w);
  r[12] = f2bfu(b1.x); r[13] = f2bfu(b1.y); r[14] = f2bfu(b1.z); r[15] = f2bfu(b1.w);
  return __builtin_bit_cast(v16bf, r);
}

// Raw LDS byte offset of a __shared__ object (LDS aperture keeps the offset in
// the low 32 bits of the generic address).
__device__ __forceinline__ unsigned ldsoff(const void* p) {
  return (unsigned)(uintptr_t)p;
}

// Async 16B global->LDS copy (ASYNCcnt-tracked, CDNA5 §10.7/§15.18).
__device__ __forceinline__ void async_b128(const unsigned short* g, unsigned lds_byte) {
  asm volatile("global_load_async_to_lds_b128 %0, %1, off"
               :: "v"(lds_byte), "v"((unsigned long long)(uintptr_t)g)
               : "memory");
}
__device__ __forceinline__ void wait_async2() {
  asm volatile("s_wait_asynccnt 0x2" ::: "memory");
}
__device__ __forceinline__ void wait_async1() {
  asm volatile("s_wait_asynccnt 0x1" ::: "memory");
}
__device__ __forceinline__ void wait_async0() {
  asm volatile("s_wait_asynccnt 0x0" ::: "memory");
}

// ---------------------------------------------------------------------------
// Kernel 0: one-shot fp32 -> bf16 conversion (8 elems/thread).
// ---------------------------------------------------------------------------
__global__ __launch_bounds__(256)
void cvt_bf16_kernel(const float* __restrict__ src, unsigned short* __restrict__ dst) {
  const size_t i = ((size_t)blockIdx.x * 256 + threadIdx.x) * 8;
  float4 f0 = *(const float4*)(src + i);
  float4 f1 = *(const float4*)(src + i + 4);
  v8u16 r;
  r[0] = f2bfu(f0.x); r[1] = f2bfu(f0.y); r[2] = f2bfu(f0.z); r[3] = f2bfu(f0.w);
  r[4] = f2bfu(f1.x); r[5] = f2bfu(f1.y); r[6] = f2bfu(f1.z); r[7] = f2bfu(f1.w);
  *(v8u16*)(dst + i) = r;
}

// ---------------------------------------------------------------------------
// Kernel 1: QKV projections, C = X @ W^T + b, bf16 sources, async double-
// buffered LDS staging.  Q/K written head-split; V written transposed [bh,d,s].
// grid = (16 ntiles, 64 mtiles, 3).
// ---------------------------------------------------------------------------
__global__ __launch_bounds__(256)
void proj_qkv_kernel(const unsigned short* __restrict__ xq,
                     const unsigned short* __restrict__ xk,
                     const unsigned short* __restrict__ xv,
                     const unsigned short* __restrict__ wqb,
                     const unsigned short* __restrict__ wkb,
                     const unsigned short* __restrict__ wvb,
                     const float* __restrict__ bq, const float* __restrict__ bk,
                     const float* __restrict__ bv,
                     unsigned short* __restrict__ qh, unsigned short* __restrict__ kh,
                     unsigned short* __restrict__ vt) {
  const unsigned short* X; const unsigned short* W; const float* B; unsigned short* O;
  switch (blockIdx.z) {
    case 0:  X = xq; W = wqb; B = bq; O = qh; break;
    case 1:  X = xk; W = wkb; B = bk; O = kh; break;
    default: X = xv; W = wvb; B = bv; O = vt; break;
  }
  __shared__ unsigned short lA[2][64 * 40];
  __shared__ unsigned short lB[2][64 * 40];

  const int tid  = threadIdx.x;
  const int m0   = blockIdx.y * 64;
  const int n0   = blockIdx.x * 64;
  const int srow = tid >> 2;           // staging row 0..63
  const int skk  = (tid & 3) * 8;      // staging k offset {0,8,16,24}
  const int lane = tid & 31;
  const int hf   = lane >> 4;
  const int kb   = hf * 8;
  const int ln   = lane & 15;
  const int wid  = tid >> 5;
  const int m16  = (wid & 3) * 16;
  const int n32  = (wid >> 2) * 32;

  const unsigned short* Asrc = X + (size_t)(m0 + srow) * 1024 + skk;
  const unsigned short* Bsrc = W + (size_t)(n0 + srow) * 1024 + skk;
  const unsigned aoff0 = ldsoff(&lA[0][srow * 40 + skk]);
  const unsigned aoff1 = ldsoff(&lA[1][srow * 40 + skk]);
  const unsigned boff0 = ldsoff(&lB[0][srow * 40 + skk]);
  const unsigned boff1 = ldsoff(&lB[1][srow * 40 + skk]);

  v8f acc0 = {}; v8f acc1 = {};

  async_b128(Asrc, aoff0);
  async_b128(Bsrc, boff0);

  for (int it = 0; it < 32; ++it) {
    const int cur = it & 1;
    if (it + 1 < 32) {
      async_b128(Asrc + (it + 1) * 32, cur ? aoff0 : aoff1);
      async_b128(Bsrc + (it + 1) * 32, cur ? boff0 : boff1);
      wait_async2();
    } else {
      wait_async0();
    }
    __syncthreads();
    v16bf a   = frag_u16(&lA[cur][(m16 + ln) * 40], kb);
    v16bf bf0 = frag_u16(&lB[cur][(n32 + ln) * 40], kb);
    v16bf bf1 = frag_u16(&lB[cur][(n32 + 16 + ln) * 40], kb);
    acc0 = WMMA_BF16(a, bf0, acc0);
    acc1 = WMMA_BF16(a, bf1, acc1);
    __syncthreads();
  }

  const bool vmode = (blockIdx.z == 2);
#pragma unroll
  for (int t = 0; t < 2; ++t) {
    v8f acc = t ? acc1 : acc0;
    const int ncol = n0 + n32 + t * 16 + ln;
    const float bias = B[ncol];
    const int h = ncol >> 6, dd = ncol & 63;
#pragma unroll
    for (int r = 0; r < 8; ++r) {
      const int mrow = m0 + m16 + r + 8 * hf;
      const int bb = mrow >> 10, ss = mrow & 1023;
      const unsigned short val = f2bfu(acc[r] + bias);
      if (vmode)
        O[(((size_t)((bb * 16 + h) * 64 + dd)) << 10) + ss] = val;   // [bh,d,s]
      else
        O[(((size_t)(bb * 16 + h) << 10) + ss) * 64 + dd] = val;     // [bh,s,d]
    }
  }
}

// ---------------------------------------------------------------------------
// Kernel 2: attention logits + fused softmax.  One block = one (b,h), a 16-row
// query tile vs all 1024 keys.  8 waves x 8 v8f accumulators.
// grid = (64 qtiles, 64 bh).
// ---------------------------------------------------------------------------
__global__ __launch_bounds__(256)
void attn_softmax_kernel(const unsigned short* __restrict__ qh,
                         const unsigned short* __restrict__ kh,
                         const float* __restrict__ mask,
                         const float* __restrict__ adjoin,
                         float* __restrict__ P) {
  const int bh  = blockIdx.y;
  const int b   = bh >> 4;
  const int q0  = blockIdx.x * 16;
  const int tid = threadIdx.x;
  const int wid = tid >> 5;
  const int lane = tid & 31;
  const int hf  = lane >> 4;
  const int kb  = hf * 8;
  const int ln  = lane & 15;
  const int n0w = wid * 128;

  v8f acc[8] = {};
  const unsigned short* qrow = qh + ((size_t)(bh << 10) + q0 + ln) * 64;

#pragma unroll
  for (int k0 = 0; k0 < 64; k0 += 32) {
    v16bf a = frag_u16(qrow + k0, kb);
#pragma unroll
    for (int t = 0; t < 8; ++t) {
      const unsigned short* krow = kh + ((size_t)(bh << 10) + n0w + t * 16 + ln) * 64;
      v16bf bf = frag_u16(krow + k0, kb);
      acc[t] = WMMA_BF16(a, bf, acc[t]);
    }
  }

#pragma unroll
  for (int t = 0; t < 8; ++t) {
    const int col = n0w + t * 16 + ln;
    const float mk = mask[b * 1024 + col] * -1e9f;
#pragma unroll
    for (int r = 0; r < 8; ++r) {
      const int qr = q0 + r + 8 * hf;
      const float adj = adjoin[((size_t)(b << 10) + qr) * 1024 + col];
      acc[t][r] = acc[t][r] * 0.125f + mk + adj;
    }
  }

  __shared__ float red[8][16];
  float rmax[8], rinv[8];

#pragma unroll
  for (int r = 0; r < 8; ++r) {
    float m = acc[0][r];
#pragma unroll
    for (int t = 1; t < 8; ++t) m = fmaxf(m, acc[t][r]);
#pragma unroll
    for (int s = 1; s < 16; s <<= 1) m = fmaxf(m, __shfl_xor(m, s, 32));
    if (ln == r) red[wid][r + 8 * hf] = m;
  }
  __syncthreads();
#pragma unroll
  for (int r = 0; r < 8; ++r) {
    float m = red[0][r + 8 * hf];
#pragma unroll
    for (int w = 1; w < 8; ++w) m = fmaxf(m, red[w][r + 8 * hf]);
    rmax[r] = m;
  }
  __syncthreads();

#pragma unroll
  for (int r = 0; r < 8; ++r) {
    float s = 0.f;
#pragma unroll
    for (int t = 0; t < 8; ++t) {
      float p = __expf(acc[t][r] - rmax[r]);
      acc[t][r] = p;
      s += p;
    }
#pragma unroll
    for (int st = 1; st < 16; st <<= 1) s += __shfl_xor(s, st, 32);
    if (ln == r) red[wid][r + 8 * hf] = s;
  }
  __syncthreads();
#pragma unroll
  for (int r = 0; r < 8; ++r) {
    float s = 0.f;
#pragma unroll
    for (int w = 0; w < 8; ++w) s += red[w][r + 8 * hf];
    rinv[r] = 1.f / s;
  }

#pragma unroll
  for (int t = 0; t < 8; ++t) {
    const int col = n0w + t * 16 + ln;
#pragma unroll
    for (int r = 0; r < 8; ++r) {
      const int qr = q0 + r + 8 * hf;
      P[((size_t)(bh << 10) + qr) * 1024 + col] = acc[t][r] * rinv[r];
    }
  }
}

// ---------------------------------------------------------------------------
// Kernel 3: context = P @ V per (b,h).  P re-read fp32 from d_out (bf16 in-
// register); transposed V staged via async double-buffered LDS.
// grid = (16 qtiles of 64, 64 bh).
// ---------------------------------------------------------------------------
__global__ __launch_bounds__(256)
void attn_pv_kernel(const float* __restrict__ P,
                    const unsigned short* __restrict__ vt,
                    unsigned short* __restrict__ ctx) {
  const int bh  = blockIdx.y;
  const int q0  = blockIdx.x * 64;
  const int tid = threadIdx.x;
  const int wid = tid >> 5, lane = tid & 31;
  const int hf  = lane >> 4, kb = hf * 8, ln = lane & 15;
  const int m16 = (wid & 3) * 16;
  const int n32 = (wid >> 2) * 32;
  const int srow = tid >> 2, skk = (tid & 3) * 8;

  __shared__ unsigned short lV[2][64 * 40];
  v8f acc0 = {}, acc1 = {};

  const unsigned short* Vsrc = vt + ((size_t)bh * 64 + srow) * 1024 + skk;
  const unsigned voff0 = ldsoff(&lV[0][srow * 40 + skk]);
  const unsigned voff1 = ldsoff(&lV[1][srow * 40 + skk]);
  const float* prow0 = P + (((size_t)bh << 10) + q0 + m16 + ln) * 1024;

  async_b128(Vsrc, voff0);

  for (int it = 0; it < 32; ++it) {
    const int cur = it & 1;
    if (it + 1 < 32) {
      async_b128(Vsrc + (it + 1) * 32, cur ? voff0 : voff1);
      wait_async1();
    } else {
      wait_async0();
    }
    __syncthreads();
    const float* prow = prow0 + it * 32;
    if (it + 1 < 32) __builtin_prefetch(prow + 32, 0, 3);
    v16bf a   = frag_f32(prow, kb);
    v16bf bf0 = frag_u16(&lV[cur][(n32 + ln) * 40], kb);
    v16bf bf1 = frag_u16(&lV[cur][(n32 + 16 + ln) * 40], kb);
    acc0 = WMMA_BF16(a, bf0, acc0);
    acc1 = WMMA_BF16(a, bf1, acc1);
    __syncthreads();
  }

  const int b = bh >> 4, h = bh & 15;
#pragma unroll
  for (int t = 0; t < 2; ++t) {
    v8f acc = t ? acc1 : acc0;
    const int dd = n32 + t * 16 + ln;
#pragma unroll
    for (int r = 0; r < 8; ++r) {
      const int qr = q0 + m16 + r + 8 * hf;
      ctx[((size_t)(b << 10) + qr) * 1024 + (h << 6) + dd] = f2bfu(acc[r]);
    }
  }
}

// ---------------------------------------------------------------------------
// Kernel 4: out = ctx @ dense_w^T + dense_b (fp32 out).  bf16 weights staged
// async; ctx fragments loaded directly from global bf16.
// grid = (16 ntiles, 64 mtiles).
// ---------------------------------------------------------------------------
__global__ __launch_bounds__(256)
void dense_kernel(const unsigned short* __restrict__ ctx,
                  const unsigned short* __restrict__ dwb,
                  const float* __restrict__ bias,
                  float* __restrict__ out) {
  const int tid = threadIdx.x;
  const int m0  = blockIdx.y * 64, n0 = blockIdx.x * 64;
  const int wid = tid >> 5, lane = tid & 31;
  const int hf  = lane >> 4, kb = hf * 8, ln = lane & 15;
  const int m16 = (wid & 3) * 16, n32 = (wid >> 2) * 32;
  const int srow = tid >> 2, skk = (tid & 3) * 8;

  __shared__ unsigned short lB[2][64 * 40];
  v8f acc0 = {}, acc1 = {};

  const unsigned short* Bsrc = dwb + (size_t)(n0 + srow) * 1024 + skk;
  const unsigned boff0 = ldsoff(&lB[0][srow * 40 + skk]);
  const unsigned boff1 = ldsoff(&lB[1][srow * 40 + skk]);
  const unsigned short* arow = ctx + (size_t)(m0 + m16 + ln) * 1024;

  async_b128(Bsrc, boff0);

  for (int it = 0; it < 32; ++it) {
    const int cur = it & 1;
    if (it + 1 < 32) {
      async_b128(Bsrc + (it + 1) * 32, cur ? boff0 : boff1);
      wait_async1();
    } else {
      wait_async0();
    }
    __syncthreads();
    v16bf a   = frag_u16(arow + it * 32, kb);
    v16bf bf0 = frag_u16(&lB[cur][(n32 + ln) * 40], kb);
    v16bf bf1 = frag_u16(&lB[cur][(n32 + 16 + ln) * 40], kb);
    acc0 = WMMA_BF16(a, bf0, acc0);
    acc1 = WMMA_BF16(a, bf1, acc1);
    __syncthreads();
  }

#pragma unroll
  for (int t = 0; t < 2; ++t) {
    v8f acc = t ? acc1 : acc0;
    const int N = n0 + n32 + t * 16 + ln;
    const float bb = bias[N];
#pragma unroll
    for (int r = 0; r < 8; ++r) {
      const int M = m0 + m16 + r + 8 * hf;
      out[(size_t)M * 1024 + N] = acc[r] + bb;
    }
  }
}

// ---------------------------------------------------------------------------
extern "C" void kernel_launch(void* const* d_in, const int* in_sizes, int n_in,
                              void* d_out, int out_size, void* d_ws, size_t ws_size,
                              hipStream_t stream) {
  const float* q      = (const float*)d_in[0];
  const float* k      = (const float*)d_in[1];
  const float* v      = (const float*)d_in[2];
  const float* mask   = (const float*)d_in[3];
  const float* adjoin = (const float*)d_in[4];
  const float* wq_w   = (const float*)d_in[5];
  const float* wq_b   = (const float*)d_in[6];
  const float* wk_w   = (const float*)d_in[7];
  const float* wk_b   = (const float*)d_in[8];
  const float* wv_w   = (const float*)d_in[9];
  const float* wv_b   = (const float*)d_in[10];
  const float* dw     = (const float*)d_in[11];
  const float* db     = (const float*)d_in[12];

  float* out = (float*)d_out;                       // [4096,1024] fp32
  float* P   = out + (size_t)4 * 1024 * 1024;       // [4,16,1024,1024] fp32

  const size_t M4 = (size_t)4 * 1024 * 1024;
  const size_t M1 = (size_t)1024 * 1024;
  unsigned short* ws  = (unsigned short*)d_ws;      // needs 64 MB
  unsigned short* xqb = ws;
  unsigned short* xkb = ws + M4;
  unsigned short* xvb = ws + 2 * M4;
  unsigned short* wqb = ws + 3 * M4;
  unsigned short* wkb = ws + 3 * M4 + M1;
  unsigned short* wvb = ws + 3 * M4 + 2 * M1;
  unsigned short* dwb = ws + 3 * M4 + 3 * M1;
  unsigned short* qh  = ws + 4 * M4;
  unsigned short* kh  = ws + 5 * M4;
  unsigned short* vt  = ws + 6 * M4;
  unsigned short* ctx = ws + 7 * M4;

  // fp32 -> bf16 pre-pass (one conversion per element, ever)
  cvt_bf16_kernel<<<2048, 256, 0, stream>>>(q, xqb);
  cvt_bf16_kernel<<<2048, 256, 0, stream>>>(k, xkb);
  cvt_bf16_kernel<<<2048, 256, 0, stream>>>(v, xvb);
  cvt_bf16_kernel<<<512, 256, 0, stream>>>(wq_w, wqb);
  cvt_bf16_kernel<<<512, 256, 0, stream>>>(wk_w, wkb);
  cvt_bf16_kernel<<<512, 256, 0, stream>>>(wv_w, wvb);
  cvt_bf16_kernel<<<512, 256, 0, stream>>>(dw, dwb);

  proj_qkv_kernel<<<dim3(16, 64, 3), 256, 0, stream>>>(
      xqb, xkb, xvb, wqb, wkb, wvb, wq_b, wk_b, wv_b, qh, kh, vt);
  attn_softmax_kernel<<<dim3(64, 64), 256, 0, stream>>>(qh, kh, mask, adjoin, P);
  attn_pv_kernel<<<dim3(16, 64), 256, 0, stream>>>(P, vt, ctx);
  dense_kernel<<<dim3(16, 64), 256, 0, stream>>>(ctx, dwb, db, out);

  (void)in_sizes; (void)n_in; (void)out_size; (void)ws_size;
}
